// LocalGlobalAttention_1443109012037
// MI455X (gfx1250) — matrix operand order repaired and checked by reading, not compiled, not run on hardware
//
#include <hip/hip_runtime.h>

typedef __bf16 bf16_t;
typedef __attribute__((ext_vector_type(16))) __bf16 v16bf;
typedef __attribute__((ext_vector_type(8)))  __bf16 v8bf;
typedef __attribute__((ext_vector_type(8)))  float  v8f;
typedef __attribute__((ext_vector_type(4)))  unsigned int u32x4;
typedef __attribute__((ext_vector_type(8)))  int i32x8;
typedef __attribute__((ext_vector_type(4)))  int i32x4;

#define NEG_BIG (-3.0e38f)

static __device__ __forceinline__ v8f wmma_bf16(v16bf a, v16bf b, v8f c) {
  // D = A(16x32 bf16) * B(32x16 bf16) + C(16x16 f32)
  return __builtin_amdgcn_wmma_f32_16x16x32_bf16(false, a, false, b, (short)0, c,
                                                 false, false);
}

static __device__ __forceinline__ v8f zero8() {
  v8f z = {0.f, 0.f, 0.f, 0.f, 0.f, 0.f, 0.f, 0.f};
  return z;
}

union V16U { v16bf v; v8bf h[2]; };

static __device__ __forceinline__ v16bf ld_pair(const bf16_t* p0, const bf16_t* p1) {
  V16U u;
  u.h[0] = *(const v8bf*)p0;
  u.h[1] = *(const v8bf*)p1;
  return u.v;
}

// --------------------------------------------------------------- TDM (CDNA5)
// 2D tensor_load_to_lds: tile_y rows of tile_x bf16 elements, row stride
// `stride_elems` in global memory, written to LDS rows padded by
// pad_amount(code) every pad_interval(code) per D# spec (08_async_tensor §8).
// tensor dims == tile dims -> no OOB clipping. cpol = 0.
// This toolchain exposes the 6-arg builtin (extra i32x8 descriptor words).
static __device__ __forceinline__ void tdm_load_2d_bf16(
    unsigned lds_byte_addr, const void* gaddr, unsigned tile_x, unsigned tile_y,
    unsigned stride_elems, unsigned pad_interval_code, unsigned pad_amount_code) {
  unsigned long long ga = (unsigned long long)(uintptr_t)gaddr;
  // group0: count=1 | lds_addr | global_addr[56:0] | type=2
  u32x4 g0;
  g0.x = 1u;
  g0.y = lds_byte_addr;
  g0.z = (unsigned)(ga & 0xffffffffu);
  g0.w = (unsigned)((ga >> 32) & 0x01ffffffu) | (2u << 30);
  // group1
  unsigned w0 = (1u << 16)                       // data_size = 2 bytes
              | (1u << 20)                       // pad_enable
              | (pad_interval_code << 22)
              | (pad_amount_code << 25);
  i32x8 g1;
  g1[0] = (int)w0;
  g1[1] = (int)((tile_x & 0xffffu) << 16);                       // tensor_dim0 lo
  g1[2] = (int)(((tile_x >> 16) & 0xffffu) |
                ((tile_y & 0xffffu) << 16));                     // td0 hi | td1 lo
  g1[3] = (int)(((tile_y >> 16) & 0xffffu) |
                ((tile_x & 0xffffu) << 16));                     // td1 hi | tile_dim0
  g1[4] = (int)(tile_y & 0xffffu);                               // tile_dim1, tile_dim2=0
  g1[5] = (int)stride_elems;                                     // tensor_dim0_stride lo32
  g1[6] = 0;                                                     // stride0 hi | stride1 lo
  g1[7] = 0;                                                     // stride1 hi
  i32x4 gz4 = {0, 0, 0, 0};
  i32x8 gz8 = {0, 0, 0, 0, 0, 0, 0, 0};
  __builtin_amdgcn_tensor_load_to_lds(g0, g1, gz4, gz4, gz8, 0);
}

static __device__ __forceinline__ unsigned lds_addr_of(const void* p) {
  return (unsigned)(uintptr_t)p;  // flat LDS aperture: offset in low 32 bits
}

// ---------------------------------------------------------------- pack kernels
__global__ void cvt_f32_bf16(const float* __restrict__ src, bf16_t* __restrict__ dst,
                             int n) {
  int i = blockIdx.x * blockDim.x + threadIdx.x;
  if (i < n) dst[i] = (bf16_t)src[i];
}

// dst[C x R] = transpose(src[R x C]) with f32 -> bf16
__global__ void pack_transpose(const float* __restrict__ src, bf16_t* __restrict__ dst,
                               int R, int C) {
  int i = blockIdx.x * blockDim.x + threadIdx.x;
  if (i >= R * C) return;
  int r = i / C, c = i - r * C;
  dst[c * R + r] = (bf16_t)src[i];
}

// ------------------------------------------------------------------- WMMA GEMM
// C[M,N](ldc) = A[M,K] * B[K,N] + bias, optional ReLU; A,B bf16 row-major.
// Block tile 128x64, 8 waves (4 along M x 2 along N), each wave 32x32 (2x2 tiles).
// A tile staged by TDM (rows 64B + 16B pad -> ld 40); B tile manual transpose.
template <bool RELU, typename OutT>
__global__ __launch_bounds__(256) void gemm_bf16(
    const bf16_t* __restrict__ A, const bf16_t* __restrict__ B,
    const float* __restrict__ bias, OutT* __restrict__ C,
    int M, int N, int K, int ldc) {
  __shared__ bf16_t sA[128 * 40];  // [row][k], TDM-padded rows (80B, 16B aligned)
  __shared__ bf16_t sB[64 * 40];   // [n][k] column-major of the 32x64 tile

  const int lane  = threadIdx.x & 31;
  const int wave  = threadIdx.x >> 5;
  const int waveM = wave & 3;
  const int waveN = wave >> 2;
  const int rowBase = blockIdx.y * 128;
  const int colBase = blockIdx.x * 64;
  const int r16  = lane & 15;
  const int half = lane >> 4;
  const int kh   = half * 8;  // A-fragment K sub-offset

  v8f acc[2][2];
#pragma unroll
  for (int mi = 0; mi < 2; ++mi)
#pragma unroll
    for (int ni = 0; ni < 2; ++ni) acc[mi][ni] = zero8();

  const int bk = threadIdx.x >> 3;        // 0..31  : B k row
  const int bn = (threadIdx.x & 7) * 8;   // B col chunk
  const unsigned sA_lds = lds_addr_of(&sA[0]);

  for (int k0 = 0; k0 < K; k0 += 32) {
    if (wave == 0) {  // A tile 128x32 via Tensor Data Mover
      tdm_load_2d_bf16(sA_lds, A + (size_t)rowBase * K + k0,
                       /*tile_x=*/32, /*tile_y=*/128, /*stride=*/(unsigned)K,
                       /*pad_interval(64B)=*/3, /*pad_amount(16B)=*/3);
    }
    {  // B tile 32x64, transposed into sB[n][k]
      const bf16_t* gp = B + (size_t)(k0 + bk) * N + (colBase + bn);
      v8bf bv = *(const v8bf*)gp;
#pragma unroll
      for (int j = 0; j < 8; ++j) sB[(bn + j) * 40 + bk] = bv[j];
    }
    if (wave == 0) __builtin_amdgcn_s_wait_tensorcnt(0);
    __syncthreads();

    v16bf af[2], bfr[2];
#pragma unroll
    for (int mi = 0; mi < 2; ++mi) {
      const bf16_t* p = &sA[(waveM * 32 + mi * 16 + r16) * 40 + kh];
      af[mi] = ld_pair(p, p + 16);  // K {kh..kh+7, 16+kh..16+kh+7}
    }
#pragma unroll
    for (int ni = 0; ni < 2; ++ni) {
      const bf16_t* p = &sB[(waveN * 32 + ni * 16 + r16) * 40 + kh];
      bfr[ni] = ld_pair(p, p + 16);  // K {kh..kh+7, 16+kh..16+kh+7}
    }
#pragma unroll
    for (int mi = 0; mi < 2; ++mi)
#pragma unroll
      for (int ni = 0; ni < 2; ++ni)
        acc[mi][ni] = wmma_bf16(af[mi], bfr[ni], acc[mi][ni]);
    __syncthreads();
  }

#pragma unroll
  for (int mi = 0; mi < 2; ++mi) {
#pragma unroll
    for (int ni = 0; ni < 2; ++ni) {
      int col = colBase + waveN * 32 + ni * 16 + r16;
      float bv = bias ? bias[col] : 0.f;
      int rb = rowBase + waveM * 32 + mi * 16 + half * 8;
#pragma unroll
      for (int i = 0; i < 8; ++i) {
        float v = acc[mi][ni][i] + bv;
        if (RELU) v = fmaxf(v, 0.f);
        C[(size_t)(rb + i) * ldc + col] = (OutT)v;
      }
    }
  }
}

// ------------------------------------------------------------------- attention
// qkv: bf16 [B*S, 1536] (q|k|v each E=512, head h at h*64). out: bf16 [B*S,512].
// One block per (b,h, 128-query slab); 8 waves, each owns a 16-row query tile.
// K tile staged by TDM (rows 128B + 16B pad -> ld 72); V manual transpose.
__global__ __launch_bounds__(256) void attn_kernel(
    const bf16_t* __restrict__ qkv, bf16_t* __restrict__ out, int is_local) {
  constexpr int S = 1024, QLD = 1536, E = 512, DH = 64, LOCAL = 3;
  __shared__ bf16_t sK[64 * 72];      // [key][dh]  (TDM-padded rows, 144B)
  __shared__ bf16_t sV[64 * 72];      // [dh][key]  (transposed for B-frags)
  __shared__ bf16_t sP[8][16 * 72];   // per-wave P scratch (A-frag relayout)

  const int lane = threadIdx.x & 31;
  const int wave = threadIdx.x >> 5;
  const int b  = blockIdx.x >> 3;
  const int h  = blockIdx.x & 7;
  const int q0 = blockIdx.y * 128;
  const int qw = q0 + wave * 16;
  const int r16  = lane & 15;
  const int half = lane >> 4;
  const int kh   = half * 8;
  const unsigned sK_lds = lds_addr_of(&sK[0]);

  // Q fragments for this wave's 16 rows (two dh-chunks of 32)
  const bf16_t* qb = qkv + (size_t)(b * S + qw + r16) * QLD + h * DH;
  v16bf qf[2];
#pragma unroll
  for (int c = 0; c < 2; ++c) {
    const bf16_t* p = qb + c * 32 + kh;
    qf[c] = ld_pair(p, p + 16);
  }

  float m[8], l[8];
#pragma unroll
  for (int i = 0; i < 8; ++i) { m[i] = NEG_BIG; l[i] = 0.f; }
  v8f o[4];
#pragma unroll
  for (int t = 0; t < 4; ++t) o[t] = zero8();

  const int key = threadIdx.x >> 2;        // 0..63
  const int ch  = (threadIdx.x & 3) * 16;  // dh chunk

  for (int kb = 0; kb < S / 64; ++kb) {
    const int ks = kb * 64;
    // uniform block-level band skip for local attention
    if (is_local && (ks > q0 + 127 + LOCAL || ks + 63 < q0 - LOCAL)) continue;

    if (wave == 0) {  // K tile 64x64 -> sK[key][dh] via TDM
      tdm_load_2d_bf16(sK_lds,
                       qkv + (size_t)(b * S + ks) * QLD + E + h * DH,
                       /*tile_x=*/64, /*tile_y=*/64, /*stride=*/QLD,
                       /*pad_interval(128B)=*/4, /*pad_amount(16B)=*/3);
    }
    {  // V tile -> sV[dh][key]
      const bf16_t* gp = qkv + (size_t)(b * S + ks + key) * QLD + 2 * E + h * DH + ch;
      v8bf v0 = *(const v8bf*)gp;
      v8bf v1 = *(const v8bf*)(gp + 8);
#pragma unroll
      for (int j = 0; j < 8; ++j) sV[(ch + j) * 72 + key] = v0[j];
#pragma unroll
      for (int j = 0; j < 8; ++j) sV[(ch + 8 + j) * 72 + key] = v1[j];
    }
    if (wave == 0) __builtin_amdgcn_s_wait_tensorcnt(0);
    __syncthreads();

    bool active = true;
    if (is_local) active = !(ks > qw + 15 + LOCAL || ks + 63 < qw - LOCAL);

    if (active) {
      // S = Q * K^T / sqrt(dh): 4 key subtiles x 2 dh chunks
      v8f sc[4];
#pragma unroll
      for (int c = 0; c < 4; ++c) {
        v8f z = zero8();
#pragma unroll
        for (int kc = 0; kc < 2; ++kc) {
          const bf16_t* p = &sK[(c * 16 + r16) * 72 + kc * 32 + kh];
          v16bf kf = ld_pair(p, p + 16);
          z = wmma_bf16(qf[kc], kf, z);
        }
        sc[c] = z * 0.125f;
      }
      if (is_local) {
#pragma unroll
        for (int c = 0; c < 4; ++c) {
          int kidx = ks + c * 16 + r16;
#pragma unroll
          for (int i = 0; i < 8; ++i) {
            int d = (qw + i + half * 8) - kidx;
            if (d > LOCAL || d < -LOCAL) sc[c][i] = NEG_BIG;
          }
        }
      }
      // online softmax (rows live across VGPR index; cols across 16 lanes)
#pragma unroll
      for (int i = 0; i < 8; ++i) {
        float rv = fmaxf(fmaxf(sc[0][i], sc[1][i]), fmaxf(sc[2][i], sc[3][i]));
#pragma unroll
        for (int off = 1; off < 16; off <<= 1)
          rv = fmaxf(rv, __shfl_xor(rv, off, 16));
        float mn = fmaxf(m[i], rv);
        float scale = (m[i] > -1.0e37f) ? __expf(m[i] - mn) : 0.f;
        float rs = 0.f;
#pragma unroll
        for (int c = 0; c < 4; ++c) {
          float s = sc[c][i];
          float p = (s > -1.0e37f) ? __expf(s - mn) : 0.f;
          sc[c][i] = p;
          rs += p;
        }
#pragma unroll
        for (int off = 1; off < 16; off <<= 1) rs += __shfl_xor(rs, off, 16);
        l[i] = l[i] * scale + rs;
        m[i] = mn;
#pragma unroll
        for (int t = 0; t < 4; ++t) o[t][i] *= scale;
      }
      // relayout P (C/D layout -> A layout) through per-wave LDS scratch
      bf16_t* ps = &sP[wave][0];
#pragma unroll
      for (int c = 0; c < 4; ++c)
#pragma unroll
        for (int i = 0; i < 8; ++i)
          ps[(i + half * 8) * 72 + c * 16 + r16] = (bf16_t)sc[c][i];
      asm volatile("s_wait_dscnt 0" ::: "memory");
      // O += P * V
#pragma unroll
      for (int kc = 0; kc < 2; ++kc) {
        const bf16_t* pa = ps + r16 * 72 + kc * 32 + kh;
        v16bf pf = ld_pair(pa, pa + 16);
#pragma unroll
        for (int t = 0; t < 4; ++t) {
          const bf16_t* pv = &sV[(t * 16 + r16) * 72 + kc * 32 + kh];
          v16bf vf = ld_pair(pv, pv + 16);
          o[t] = wmma_bf16(pf, vf, o[t]);
        }
      }
    }
    __syncthreads();
  }

#pragma unroll
  for (int i = 0; i < 8; ++i) l[i] = (l[i] > 0.f) ? 1.f / l[i] : 0.f;
#pragma unroll
  for (int t = 0; t < 4; ++t) {
#pragma unroll
    for (int i = 0; i < 8; ++i) {
      float v = o[t][i] * l[i];
      int row = qw + i + half * 8;
      int col = h * DH + t * 16 + r16;
      out[(size_t)(b * S + row) * E + col] = (bf16_t)v;
    }
  }
}

// --------------------------------------------------------------------- launch
extern "C" void kernel_launch(void* const* d_in, const int* in_sizes, int n_in,
                              void* d_out, int out_size, void* d_ws, size_t ws_size,
                              hipStream_t stream) {
  (void)in_sizes; (void)n_in; (void)out_size; (void)ws_size;
  const float* x      = (const float*)d_in[0];
  const float* Wl_in  = (const float*)d_in[1];
  const float* bl_in  = (const float*)d_in[2];
  const float* Wl_out = (const float*)d_in[3];
  const float* bl_out = (const float*)d_in[4];
  const float* Wg_in  = (const float*)d_in[5];
  const float* bg_in  = (const float*)d_in[6];
  const float* Wg_out = (const float*)d_in[7];
  const float* bg_out = (const float*)d_in[8];
  const float* Wf     = (const float*)d_in[9];
  const float* bfv    = (const float*)d_in[10];
  float* outp = (float*)d_out;

  const int B = 8, S = 1024, E = 512;
  const int M = B * S;  // 8192
  char* ws = (char*)d_ws;
  size_t off = 0;
  auto alloc = [&](size_t bytes) -> char* {
    char* p = ws + off;
    off = (off + bytes + 255) & ~(size_t)255;
    return p;
  };
  bf16_t* xb     = (bf16_t*)alloc((size_t)M * E * 2);
  bf16_t* wlinT  = (bf16_t*)alloc((size_t)E * 3 * E * 2);
  bf16_t* wginT  = (bf16_t*)alloc((size_t)E * 3 * E * 2);
  bf16_t* wloutT = (bf16_t*)alloc((size_t)E * E * 2);
  bf16_t* wgoutT = (bf16_t*)alloc((size_t)E * E * 2);
  bf16_t* wfT    = (bf16_t*)alloc((size_t)2 * E * E * 2);
  bf16_t* qkv_l  = (bf16_t*)alloc((size_t)M * 3 * E * 2);
  bf16_t* qkv_g  = (bf16_t*)alloc((size_t)M * 3 * E * 2);
  bf16_t* attn_l = (bf16_t*)alloc((size_t)M * E * 2);
  bf16_t* attn_g = (bf16_t*)alloc((size_t)M * E * 2);
  bf16_t* fused  = (bf16_t*)alloc((size_t)M * 2 * E * 2);

  cvt_f32_bf16<<<(M * E + 255) / 256, 256, 0, stream>>>(x, xb, M * E);
  pack_transpose<<<(3 * E * E + 255) / 256, 256, 0, stream>>>(Wl_in, wlinT, 3 * E, E);
  pack_transpose<<<(3 * E * E + 255) / 256, 256, 0, stream>>>(Wg_in, wginT, 3 * E, E);
  pack_transpose<<<(E * E + 255) / 256, 256, 0, stream>>>(Wl_out, wloutT, E, E);
  pack_transpose<<<(E * E + 255) / 256, 256, 0, stream>>>(Wg_out, wgoutT, E, E);
  pack_transpose<<<(2 * E * E + 255) / 256, 256, 0, stream>>>(Wf, wfT, E, 2 * E);

  dim3 blk(256);
  gemm_bf16<false, bf16_t><<<dim3(3 * E / 64, M / 128), blk, 0, stream>>>(
      xb, wlinT, bl_in, qkv_l, M, 3 * E, E, 3 * E);
  gemm_bf16<false, bf16_t><<<dim3(3 * E / 64, M / 128), blk, 0, stream>>>(
      xb, wginT, bg_in, qkv_g, M, 3 * E, E, 3 * E);

  attn_kernel<<<dim3(B * 8, S / 128), blk, 0, stream>>>(qkv_l, attn_l, 1);
  attn_kernel<<<dim3(B * 8, S / 128), blk, 0, stream>>>(qkv_g, attn_g, 0);

  gemm_bf16<false, bf16_t><<<dim3(E / 64, M / 128), blk, 0, stream>>>(
      attn_l, wloutT, bl_out, fused, M, E, E, 2 * E);
  gemm_bf16<false, bf16_t><<<dim3(E / 64, M / 128), blk, 0, stream>>>(
      attn_g, wgoutT, bg_out, fused + E, M, E, E, 2 * E);

  gemm_bf16<true, float><<<dim3(E / 64, M / 128), blk, 0, stream>>>(
      fused, wfT, bfv, outp, M, E, 2 * E, E);
}
